// Block_9603546874374
// MI455X (gfx1250) — matrix operand order repaired and checked by reading, not compiled, not run on hardware
//
#include <hip/hip_runtime.h>
#include <hip/hip_bf16.h>

#define Dd   1024
#define Hh   16
#define HDd  64
#define FFf  4096
#define Bb   8
#define Nn   1024
#define Mm   (Bb*Nn)   // 8192 rows

typedef __attribute__((ext_vector_type(4)))  __bf16 v4bf;
typedef __attribute__((ext_vector_type(8)))  __bf16 v8bf;
typedef __attribute__((ext_vector_type(16))) __bf16 v16bf;
typedef __attribute__((ext_vector_type(8)))  float  v8f;

static __device__ inline v8f zero8() {
  v8f z = {0.f,0.f,0.f,0.f,0.f,0.f,0.f,0.f};
  return z;
}

static __device__ inline v8f wmma_bf16(v16bf a, v16bf b, v8f c) {
  // D = A(16x32 bf16) x B(32x16 bf16) + C(16x16 f32)
  return __builtin_amdgcn_wmma_f32_16x16x32_bf16(
      /*neg_a=*/false, a, /*neg_b=*/false, b,
      /*c_mod=*/(short)0, c, /*reuse_a=*/false, /*reuse_b=*/false);
}

// A-fragment 16x32 (M x K), 16-bit: lane row = lane&15, two 16-byte chunks at
// k0 + (lane>>4)*16 and k0 + (lane>>4)*16 + 8.
static __device__ inline v16bf load_a_frag(const __bf16* rowbase, int ld, int k0, int lane) {
  int m = lane & 15, h = lane >> 4;
  const __bf16* p = rowbase + (size_t)m * ld + k0 + h * 16;
  v8bf lo = *(const v8bf*)p;
  v8bf hi = *(const v8bf*)(p + 8);
  return __builtin_shufflevector(lo, hi, 0,1,2,3,4,5,6,7,8,9,10,11,12,13,14,15);
}

// B-fragment 32x16 (K x N) built from a row-major [N,K] matrix (i.e. B = W^T):
// lane (n = lane&15) holds 16 contiguous K values starting at k0 + (lane>>4)*16.
static __device__ inline v16bf load_b_frag(const __bf16* colbase, int ld, int k0, int lane) {
  int n = lane & 15, h = lane >> 4;
  return *(const v16bf*)(colbase + (size_t)n * ld + k0 + h * 16);
}

// Async global -> LDS copy of one 16-byte segment (tracked by ASYNCcnt).
static __device__ inline void async_a_load_b128(const __bf16* gsrc, __bf16* ldst) {
  unsigned lds_off = (unsigned)(size_t)ldst;   // low 32 bits of flat addr = LDS byte addr
  asm volatile("global_load_async_to_lds_b128 %0, %1, off"
               :: "v"(lds_off), "v"(gsrc)
               : "memory");
}

// ---------------------------------------------------------------- utilities
__global__ void cast_w_kernel(const float* __restrict__ in, __bf16* __restrict__ out, int n4) {
  int i = blockIdx.x * 256 + threadIdx.x;
  if (i < n4) {
    float4 v = ((const float4*)in)[i];
    v4bf o = { (__bf16)v.x, (__bf16)v.y, (__bf16)v.z, (__bf16)v.w };
    ((v4bf*)out)[i] = o;
  }
}

__global__ void qkv_bias_kernel(const float* __restrict__ qb, const float* __restrict__ vb,
                                float* __restrict__ out) {
  int i = blockIdx.x * 256 + threadIdx.x;
  if (i < 3 * Dd)
    out[i] = (i < Dd) ? qb[i] : ((i < 2 * Dd) ? 0.f : vb[i - 2 * Dd]);
}

// LayerNorm over D=1024, one block (256 threads, 8 waves) per row, bf16 out.
__global__ __launch_bounds__(256)
void ln_cast_kernel(const float* __restrict__ x, const float* __restrict__ g,
                    const float* __restrict__ b, __bf16* __restrict__ out) {
  int row = blockIdx.x, tid = threadIdx.x;
  float4 v = ((const float4*)(x + (size_t)row * Dd))[tid];
  float s  = v.x + v.y + v.z + v.w;
  float s2 = v.x*v.x + v.y*v.y + v.z*v.z + v.w*v.w;
  #pragma unroll
  for (int o = 16; o > 0; o >>= 1) { s += __shfl_down(s, o); s2 += __shfl_down(s2, o); }
  __shared__ float red[16];
  if ((tid & 31) == 0) { red[tid >> 5] = s; red[8 + (tid >> 5)] = s2; }
  __syncthreads();
  s = 0.f; s2 = 0.f;
  #pragma unroll
  for (int i = 0; i < 8; i++) { s += red[i]; s2 += red[8 + i]; }
  float mu  = s * (1.f / Dd);
  float var = s2 * (1.f / Dd) - mu * mu;
  float rs  = rsqrtf(var + 1e-5f);
  float4 gg = ((const float4*)g)[tid];
  float4 bb = ((const float4*)b)[tid];
  v4bf o = { (__bf16)((v.x - mu) * rs * gg.x + bb.x),
             (__bf16)((v.y - mu) * rs * gg.y + bb.y),
             (__bf16)((v.z - mu) * rs * gg.z + bb.z),
             (__bf16)((v.w - mu) * rs * gg.w + bb.w) };
  ((v4bf*)(out + (size_t)row * Dd))[tid] = o;
}

// vT[b,h,d,n] = qkv[(b*N+n)*3D + 2D + h*64 + d]  (makes attn.V B-frags contiguous)
__global__ void vtrans_kernel(const __bf16* __restrict__ qkv, __bf16* __restrict__ vT) {
  size_t idx = (size_t)blockIdx.x * 256 + threadIdx.x;   // 2^23 threads
  int n = idx & 1023;
  int d = (idx >> 10) & 63;
  int h = (idx >> 16) & 15;
  int b = (int)(idx >> 20);
  vT[idx] = qkv[((size_t)(b * Nn + n)) * (3 * Dd) + 2 * Dd + h * HDd + d];
}

// ---------------------------------------------------------------- GEMM
// C[M,Ntot] = A[M,K](bf16) * W[Ntot,K]^T(bf16) + bias, epilogues:
//   EPI 0: store bf16; EPI 1: GELU(exact) then store bf16; EPI 2: f32 + residual.
// Block 256 thr = 8 waves (2x4); wave tile 32x64 -> block tile 64x256.
// A-tile (64x64 per stage, 8 KB) is double-buffered through LDS via async
// copies (2x16B per thread); 16 WMMAs per stage between barriers.
// B streams straight from global (all bf16 weights are L2-resident: 24MB/192MB).
template <int EPI>
__global__ __launch_bounds__(256)
void gemm_kernel(const __bf16* __restrict__ A, const __bf16* __restrict__ W,
                 const float* __restrict__ bias, const float* __restrict__ residual,
                 void* __restrict__ outp, int Ntot, int K) {
  int lane = threadIdx.x & 31;
  int w    = threadIdx.x >> 5;
  int wr   = w >> 2, wc = w & 3;
  int m0b  = blockIdx.y * 64;
  int m0   = m0b + wr * 32;
  int n0   = blockIdx.x * 256 + wc * 64;

  __shared__ __align__(16) __bf16 atile[2][64 * 64];   // 16 KB double buffer

  // staging map: thread t -> row t>>2 (0..63), 32B segment (t&3)*16 elements
  int arow = threadIdx.x >> 2;
  int acol = (threadIdx.x & 3) * 16;
  const __bf16* agsrc = A + (size_t)(m0b + arow) * K + acol;
  __bf16* adst[2] = { &atile[0][arow * 64 + acol],
                      &atile[1][arow * 64 + acol] };

  v8f acc[2][4];
  #pragma unroll
  for (int i = 0; i < 2; i++)
    #pragma unroll
    for (int t = 0; t < 4; t++) acc[i][t] = zero8();

  const int nk = K / 64;
  async_a_load_b128(agsrc,     adst[0]);           // prologue: stage 0 (2x16B)
  async_a_load_b128(agsrc + 8, adst[0] + 8);

  for (int i = 0; i < nk; i++) {
    int cur = i & 1;
    if (i + 1 < nk) {
      const __bf16* g = agsrc + (size_t)(i + 1) * 64;
      async_a_load_b128(g,     adst[cur ^ 1]);
      async_a_load_b128(g + 8, adst[cur ^ 1] + 8);
      asm volatile("s_wait_asynccnt 0x2" ::: "memory");   // oldest 2 (cur stage) landed
    } else {
      asm volatile("s_wait_asynccnt 0x0" ::: "memory");
    }
    __syncthreads();                                 // publish buf[cur]

    const __bf16* ab = &atile[cur][0];
    #pragma unroll
    for (int kk = 0; kk < 2; kk++) {
      int k0 = kk * 32;
      v16bf a0 = load_a_frag(ab + (size_t)(wr * 32) * 64,      64, k0, lane);
      v16bf a1 = load_a_frag(ab + (size_t)(wr * 32 + 16) * 64, 64, k0, lane);
      int kg = i * 64 + k0;
      #pragma unroll
      for (int t = 0; t < 4; t++) {
        v16bf bf = load_b_frag(W + (size_t)(n0 + t * 16) * K, K, kg, lane);
        acc[0][t] = wmma_bf16(a0, bf, acc[0][t]);
        acc[1][t] = wmma_bf16(a1, bf, acc[1][t]);
      }
    }
    __syncthreads();                                 // all readers done before refill
  }

  int hi = lane >> 4, nn = lane & 15;
  #pragma unroll
  for (int i = 0; i < 2; i++)
    #pragma unroll
    for (int t = 0; t < 4; t++)
      #pragma unroll
      for (int r = 0; r < 8; r++) {
        int row = m0 + i * 16 + r + 8 * hi;
        int col = n0 + t * 16 + nn;
        size_t idx = (size_t)row * Ntot + col;
        float v = acc[i][t][r] + bias[col];
        if (EPI == 1) v = 0.5f * v * (1.f + erff(v * 0.70710678118f));
        if (EPI <= 1) ((__bf16*)outp)[idx] = (__bf16)v;
        else          ((float*)outp)[idx]  = residual[idx] + v;
      }
}

// ---------------------------------------------------------------- attention
static __device__ inline float red16_max(float v) {
  v = fmaxf(v, __shfl_xor(v, 1));
  v = fmaxf(v, __shfl_xor(v, 2));
  v = fmaxf(v, __shfl_xor(v, 4));
  v = fmaxf(v, __shfl_xor(v, 8));
  return v;
}
static __device__ inline float red16_sum(float v) {
  v += __shfl_xor(v, 1);
  v += __shfl_xor(v, 2);
  v += __shfl_xor(v, 4);
  v += __shfl_xor(v, 8);
  return v;
}

// grid (N/64, B*H), 128 threads = 4 waves; each wave owns 16 query rows.
__global__ __launch_bounds__(128)
void attn_kernel(const __bf16* __restrict__ qkv, const __bf16* __restrict__ vT,
                 __bf16* __restrict__ obuf) {
  int bh = blockIdx.y;
  int b = bh >> 4, h = bh & 15;
  int w = threadIdx.x >> 5, lane = threadIdx.x & 31;
  int qrow = blockIdx.x * 64 + w * 16;
  int hi = lane >> 4, nn = lane & 15;

  __shared__ __align__(32) __bf16 pstage[4][16 * 32];   // per-wave P tile (16x32)
  __bf16* pp = pstage[w];

  const __bf16* qbase = qkv + ((size_t)(b * Nn + qrow)) * (3 * Dd) + h * HDd;
  const __bf16* kbase = qkv + ((size_t)(b * Nn)) * (3 * Dd) + Dd + h * HDd;
  const __bf16* vtb   = vT + ((size_t)(b * Hh + h)) * HDd * Nn;

  v16bf qA0 = load_a_frag(qbase, 3 * Dd, 0,  lane);
  v16bf qA1 = load_a_frag(qbase, 3 * Dd, 32, lane);

  float m_s[8], l_s[8];
  v8f oac[4];
  #pragma unroll
  for (int r = 0; r < 8; r++) { m_s[r] = -1e30f; l_s[r] = 0.f; }
  #pragma unroll
  for (int t = 0; t < 4; t++) oac[t] = zero8();

  const float scale = 0.125f;   // HD^-0.5 = 1/8

  for (int j = 0; j < Nn; j += 32) {
    // scores: two 16x16 tiles, K-dim = HD = 64 (2 wmma each)
    v8f s[2];
    #pragma unroll
    for (int kb = 0; kb < 2; kb++) {
      const __bf16* kb0 = kbase + (size_t)(j + kb * 16) * (3 * Dd);
      v8f z = zero8();
      z = wmma_bf16(qA0, load_b_frag(kb0, 3 * Dd, 0,  lane), z);
      z = wmma_bf16(qA1, load_b_frag(kb0, 3 * Dd, 32, lane), z);
      s[kb] = z;
    }
    // online softmax: lane half `hi` holds rows r+8*hi for col nn
    #pragma unroll
    for (int r = 0; r < 8; r++) {
      float s0 = s[0][r] * scale, s1 = s[1][r] * scale;
      float mnew = fmaxf(m_s[r], red16_max(fmaxf(s0, s1)));
      float p0 = __expf(s0 - mnew), p1 = __expf(s1 - mnew);
      float rs = red16_sum(p0 + p1);
      float f  = __expf(m_s[r] - mnew);
      l_s[r] = l_s[r] * f + rs;
      m_s[r] = mnew;
      #pragma unroll
      for (int t = 0; t < 4; t++) oac[t][r] = oac[t][r] * f;
      pp[(r + 8 * hi) * 32 + nn]      = (__bf16)p0;
      pp[(r + 8 * hi) * 32 + 16 + nn] = (__bf16)p1;
    }
    // P (16x32) as A-fragment from LDS, then O += P * V (4 d-tiles)
    v16bf pA = load_a_frag(pp, 32, 0, lane);
    #pragma unroll
    for (int t = 0; t < 4; t++) {
      v16bf vB = load_b_frag(vtb + (size_t)(t * 16) * Nn, Nn, j, lane);
      oac[t] = wmma_bf16(pA, vB, oac[t]);
    }
  }

  // normalize + store (bf16, [8192, 1024] activation layout)
  #pragma unroll
  for (int t = 0; t < 4; t++)
    #pragma unroll
    for (int r = 0; r < 8; r++) {
      int row = qrow + r + 8 * hi;
      int col = h * HDd + t * 16 + nn;
      obuf[(size_t)(b * Nn + row) * Dd + col] = (__bf16)(oac[t][r] / l_s[r]);
    }
}

// ---------------------------------------------------------------- launcher
extern "C" void kernel_launch(void* const* d_in, const int* in_sizes, int n_in,
                              void* d_out, int out_size, void* d_ws, size_t ws_size,
                              hipStream_t stream) {
  const float* x      = (const float*)d_in[0];
  const float* ln1_g  = (const float*)d_in[1];
  const float* ln1_b  = (const float*)d_in[2];
  const float* ln2_g  = (const float*)d_in[3];
  const float* ln2_b  = (const float*)d_in[4];
  const float* qkv_w  = (const float*)d_in[5];
  const float* q_bias = (const float*)d_in[6];
  const float* v_bias = (const float*)d_in[7];
  const float* proj_w = (const float*)d_in[8];
  const float* proj_b = (const float*)d_in[9];
  const float* fc1_w  = (const float*)d_in[10];
  const float* fc1_b  = (const float*)d_in[11];
  const float* fc2_w  = (const float*)d_in[12];
  const float* fc2_b  = (const float*)d_in[13];
  float* out = (float*)d_out;

  const size_t MB = (size_t)1 << 20;
  char* ws = (char*)d_ws;
  __bf16* h_bf  = (__bf16*)(ws);              // 16 MB  (LN1 out, reused for LN2 out)
  __bf16* qkv_o = (__bf16*)(ws + 16 * MB);    // 48 MB
  __bf16* vT    = (__bf16*)(ws + 64 * MB);    // 16 MB
  __bf16* obuf  = (__bf16*)(ws + 80 * MB);    // 16 MB
  float*  x1    = (float*)(ws + 96 * MB);     // 32 MB
  __bf16* wqkv  = (__bf16*)(ws + 128 * MB);   //  6 MB
  __bf16* wproj = (__bf16*)(ws + 134 * MB);   //  2 MB
  __bf16* wfc1  = (__bf16*)(ws + 136 * MB);   //  8 MB
  __bf16* wfc2  = (__bf16*)(ws + 144 * MB);   //  8 MB
  float*  qkvb  = (float*)(ws + 152 * MB);    // 12 KB
  __bf16* hf    = (__bf16*)(ws + 16 * MB);    // 64 MB, aliases dead qkv_o+vT

  // 1) weights -> bf16, qkv bias assembly
  cast_w_kernel<<<(3 * Dd * Dd / 4 + 255) / 256, 256, 0, stream>>>(qkv_w, wqkv, 3 * Dd * Dd / 4);
  cast_w_kernel<<<(Dd * Dd / 4 + 255) / 256, 256, 0, stream>>>(proj_w, wproj, Dd * Dd / 4);
  cast_w_kernel<<<(FFf * Dd / 4 + 255) / 256, 256, 0, stream>>>(fc1_w, wfc1, FFf * Dd / 4);
  cast_w_kernel<<<(Dd * FFf / 4 + 255) / 256, 256, 0, stream>>>(fc2_w, wfc2, Dd * FFf / 4);
  qkv_bias_kernel<<<12, 256, 0, stream>>>(q_bias, v_bias, qkvb);

  // 2) LN1 -> bf16
  ln_cast_kernel<<<Mm, 256, 0, stream>>>(x, ln1_g, ln1_b, h_bf);

  // 3) QKV GEMM: [8192,1024] x [3072,1024]^T
  gemm_kernel<0><<<dim3(3 * Dd / 256, Mm / 64), 256, 0, stream>>>(
      h_bf, wqkv, qkvb, nullptr, qkv_o, 3 * Dd, Dd);

  // 4) transpose V per head
  vtrans_kernel<<<(Bb * Hh * HDd * Nn) / 256, 256, 0, stream>>>(qkv_o, vT);

  // 5) attention (flash-style online softmax)
  attn_kernel<<<dim3(Nn / 64, Bb * Hh), 128, 0, stream>>>(qkv_o, vT, obuf);

  // 6) proj GEMM + residual -> x1 (f32)
  gemm_kernel<2><<<dim3(Dd / 256, Mm / 64), 256, 0, stream>>>(
      obuf, wproj, proj_b, x, x1, Dd, Dd);

  // 7) LN2 -> bf16 (reuse h_bf)
  ln_cast_kernel<<<Mm, 256, 0, stream>>>(x1, ln2_g, ln2_b, h_bf);

  // 8) FC1 GEMM + GELU -> hf (bf16)
  gemm_kernel<1><<<dim3(FFf / 256, Mm / 64), 256, 0, stream>>>(
      h_bf, wfc1, fc1_b, nullptr, hf, FFf, Dd);

  // 9) FC2 GEMM + residual -> out (f32)
  gemm_kernel<2><<<dim3(Dd / 256, Mm / 64), 256, 0, stream>>>(
      hf, wfc2, fc2_b, x1, out, Dd, FFf);
}
